// SelfAttn_76940044140921
// MI455X (gfx1250) — matrix operand order repaired and checked by reading, not compiled, not run on hardware
//
#include <hip/hip_runtime.h>

typedef __attribute__((ext_vector_type(16))) __bf16        v16bf;
typedef __attribute__((ext_vector_type(8)))  float         v8f;
typedef __attribute__((ext_vector_type(4)))  unsigned int  u32x4;
typedef __attribute__((ext_vector_type(4)))  float         f32x4;

union FragBF {
  v16bf v;
  u32x4 q[2];
};

__device__ __forceinline__ unsigned short f2bf(float f) {
  unsigned int u = __float_as_uint(f);
  unsigned int r = u + 0x7FFFu + ((u >> 16) & 1u);   // round-to-nearest-even
  return (unsigned short)(r >> 16);
}

// pack two f32 -> packed bf16 pair (truncation), lo value in low 16 bits
__device__ __forceinline__ unsigned int pk_bf16(float flo, float fhi) {
  return __builtin_amdgcn_perm(__float_as_uint(fhi), __float_as_uint(flo),
                               0x07060302u);
}

// ---------------------------------------------------------------------------
// Pass 1: fused QKV projection.  x:[B,64,4096] f32 -> Qb:[B,4096,32] bf16
// (pre-scaled by log2(e)/sqrt(N)), Kb:[B,4096,32] bf16, Vt:[B,64,4096] bf16.
// ---------------------------------------------------------------------------
__global__ __launch_bounds__(256)
void qkv_proj_kernel(const float* __restrict__ x,
                     const float* __restrict__ wq, const float* __restrict__ bq,
                     const float* __restrict__ wk, const float* __restrict__ bk,
                     const float* __restrict__ wv, const float* __restrict__ bv,
                     unsigned short* __restrict__ Qb,
                     unsigned short* __restrict__ Kb,
                     unsigned short* __restrict__ Vt)
{
  const int N = 4096;
  __shared__ float xt[64][129];
  const int b  = blockIdx.x >> 5;          // 32 n-tiles per batch
  const int n0 = (blockIdx.x & 31) * 128;
  const int t  = threadIdx.x;

  for (int i = t; i < 64 * 128; i += 256) {
    int c = i >> 7, nl = i & 127;
    xt[c][nl] = x[((size_t)(b * 64 + c)) * N + n0 + nl];
  }
  __syncthreads();

  const int half = t >> 7;
  const int nl   = t & 127;
  const int n    = n0 + nl;
  // 1/sqrt(4096) * log2(e): softmax done in exp2 domain
  const float scale = (1.0f / 64.0f) * 1.44269504088896340736f;

  if (half == 0) {
    for (int o = 0; o < 32; ++o) {
      float aq = bq[o], ak = bk[o];
      #pragma unroll 16
      for (int c = 0; c < 64; ++c) {
        float xv = xt[c][nl];
        aq = fmaf(wq[o * 64 + c], xv, aq);
        ak = fmaf(wk[o * 64 + c], xv, ak);
      }
      Qb[((size_t)(b * N + n)) * 32 + o] = f2bf(aq * scale);
      Kb[((size_t)(b * N + n)) * 32 + o] = f2bf(ak);
    }
  } else {
    for (int o = 0; o < 64; ++o) {
      float av = bv[o];
      #pragma unroll 16
      for (int c = 0; c < 64; ++c)
        av = fmaf(wv[o * 64 + c], xt[c][nl], av);
      Vt[((size_t)(b * 64 + o)) * N + n] = f2bf(av);
    }
  }
}

// ---------------------------------------------------------------------------
// Pass 2: flash attention, transposed-S formulation, 128-key chunks.
// S^T = K * Q^T puts queries on lanes / keys on VGPR components, so
//  - softmax stats are per-lane scalars (balanced VALU tree + one xor-16 shfl)
//  - the S^T accumulator pairs ARE the bf16 A-layout of P (register-only pack)
// All per-chunk overhead (alpha exp, LDS broadcast, O-rescale) amortized
// over 128 keys.  One wave owns a 16-query tile; 4 waves per block.
// ---------------------------------------------------------------------------
__global__ __launch_bounds__(128)
void flash_attn_kernel(const unsigned short* __restrict__ Qb,
                       const unsigned short* __restrict__ Kb,
                       const unsigned short* __restrict__ Vt,
                       float* __restrict__ AO)
{
  const int N = 4096;
  __shared__ float bsc[4][16];                 // per-wave stat broadcast

  const int wave = threadIdx.x >> 5;
  const int lane = threadIdx.x & 31;
  const int lo   = lane & 15;
  const int hi   = (lane >> 4) & 1;

  const int b      = blockIdx.x >> 6;          // 64 tile-groups per batch
  const int tgroup = blockIdx.x & 63;
  const int qbase  = (tgroup * 4 + wave) * 16;

  // Q^T B-fragment (32 dims x 16 queries): lane = query col, halves split dims
  FragBF qf;
  {
    const unsigned short* qp =
        Qb + ((size_t)(b * N + qbase + lo)) * 32 + hi * 8;
    qf.q[0] = *(const u32x4*)(qp);
    qf.q[1] = *(const u32x4*)(qp + 16);
  }

  v8f Oa[4];
  #pragma unroll
  for (int c = 0; c < 4; ++c) Oa[c] = (v8f){};
  float m = -3.0e38f, L = 0.0f;

  float* asc = bsc[wave];
  const unsigned short* Kbase = Kb + (size_t)b * N * 32;
  const unsigned short* Vbase = Vt + (size_t)b * 64 * N;

  for (int kb = 0; kb < N; kb += 128) {
    // prefetch next key chunk (global_prefetch_b8)
    __builtin_prefetch(Kbase + (size_t)(kb + 128 + lo) * 32, 0, 1);

    // ---- S^T = K * Q^T : 8 key sub-tiles of 16 ----
    v8f s[8];
    #pragma unroll
    for (int j = 0; j < 8; ++j) {
      FragBF kf;  // A-fragment: lane row = key, halves split dims 0-7/16-23 vs 8-15/24-31
      const unsigned short* kp =
          Kbase + ((size_t)(kb + j * 16 + lo)) * 32 + hi * 8;
      kf.q[0] = *(const u32x4*)(kp);
      kf.q[1] = *(const u32x4*)(kp + 16);
      v8f z = (v8f){};
      s[j] = __builtin_amdgcn_wmma_f32_16x16x32_bf16(
          false, kf.v, false, qf.v, (short)0, z, false, false);
    }

    // ---- per-lane online softmax over the 64 keys this lane holds ----
    // balanced max tree: 8 independent per-subtile chains, then log combine
    float mj[8];
    #pragma unroll
    for (int j = 0; j < 8; ++j) {
      float a0 = fmaxf(s[j][0], s[j][1]);
      float a1 = fmaxf(s[j][2], s[j][3]);
      float a2 = fmaxf(s[j][4], s[j][5]);
      float a3 = fmaxf(s[j][6], s[j][7]);
      mj[j] = fmaxf(fmaxf(a0, a1), fmaxf(a2, a3));
    }
    float mx = fmaxf(fmaxf(fmaxf(mj[0], mj[1]), fmaxf(mj[2], mj[3])),
                     fmaxf(fmaxf(mj[4], mj[5]), fmaxf(mj[6], mj[7])));
    mx = fmaxf(mx, __shfl_xor(mx, 16));        // combine lane halves
    const float mn    = fmaxf(m, mx);
    const float alpha = __builtin_amdgcn_exp2f(m - mn);
    m = mn;

    // exponentials + balanced sum tree (independent per-subtile partials)
    float sj[8];
    #pragma unroll
    for (int j = 0; j < 8; ++j) {
      float p0 = __builtin_amdgcn_exp2f(s[j][0] - mn);
      float p1 = __builtin_amdgcn_exp2f(s[j][1] - mn);
      float p2 = __builtin_amdgcn_exp2f(s[j][2] - mn);
      float p3 = __builtin_amdgcn_exp2f(s[j][3] - mn);
      float p4 = __builtin_amdgcn_exp2f(s[j][4] - mn);
      float p5 = __builtin_amdgcn_exp2f(s[j][5] - mn);
      float p6 = __builtin_amdgcn_exp2f(s[j][6] - mn);
      float p7 = __builtin_amdgcn_exp2f(s[j][7] - mn);
      s[j][0] = p0; s[j][1] = p1; s[j][2] = p2; s[j][3] = p3;
      s[j][4] = p4; s[j][5] = p5; s[j][6] = p6; s[j][7] = p7;
      sj[j] = ((p0 + p1) + (p2 + p3)) + ((p4 + p5) + (p6 + p7));
    }
    float sum = ((sj[0] + sj[1]) + (sj[2] + sj[3])) +
                ((sj[4] + sj[5]) + (sj[6] + sj[7]));
    sum += __shfl_xor(sum, 16);
    L = L * alpha + sum;

    // ---- broadcast per-query alpha to O-row layout (3 DS ops / 128 keys) ----
    asc[lo] = alpha;                           // both halves write same value
    union { f32x4 h[2]; float f[8]; } av;
    av.h[0] = *(const f32x4*)(asc + hi * 8);
    av.h[1] = *(const f32x4*)(asc + hi * 8 + 4);
    #pragma unroll
    for (int c = 0; c < 4; ++c)
      #pragma unroll
      for (int v = 0; v < 8; ++v)
        Oa[c][v] *= av.f[v];

    // ---- P A-fragments: S^T accumulator pairs are already in A layout ----
    FragBF pa[4];
    #pragma unroll
    for (int h = 0; h < 4; ++h) {
      #pragma unroll
      for (int p = 0; p < 4; ++p) {
        pa[h].q[0][p] = pk_bf16(s[2 * h    ][2 * p], s[2 * h    ][2 * p + 1]);
        pa[h].q[1][p] = pk_bf16(s[2 * h + 1][2 * p], s[2 * h + 1][2 * p + 1]);
      }
    }

    // ---- O = O*alpha + P * V  (4 channel tiles x 4 key chunks of 32) ----
    #pragma unroll
    for (int c = 0; c < 4; ++c) {
      const unsigned short* vp =
          Vbase + (size_t)(c * 16 + lo) * N + kb + hi * 8;
      #pragma unroll
      for (int kk = 0; kk < 4; ++kk) {
        FragBF vf;
        vf.q[0] = *(const u32x4*)(vp + kk * 32);
        vf.q[1] = *(const u32x4*)(vp + kk * 32 + 16);
        Oa[c] = __builtin_amdgcn_wmma_f32_16x16x32_bf16(
            false, pa[kk].v, false, vf.v, (short)0, Oa[c], false, false);
      }
    }
  }

  // ---- normalize by row sums and store AO[b][ch][n] ----
  asc[lo] = L;                                 // broadcast L to row layout
  union { f32x4 h[2]; float f[8]; } lv;
  lv.h[0] = *(const f32x4*)(asc + hi * 8);
  lv.h[1] = *(const f32x4*)(asc + hi * 8 + 4);
  float inv[8];
  #pragma unroll
  for (int v = 0; v < 8; ++v) inv[v] = 1.0f / lv.f[v];

  #pragma unroll
  for (int c = 0; c < 4; ++c) {
    union { v8f v; f32x4 h[2]; } o;
    #pragma unroll
    for (int v = 0; v < 8; ++v) o.v[v] = Oa[c][v] * inv[v];
    float* dst = AO + ((size_t)(b * 64 + c * 16 + lo)) * N + qbase + hi * 8;
    *(f32x4*)(dst)     = o.h[0];
    *(f32x4*)(dst + 4) = o.h[1];
  }
}

// ---------------------------------------------------------------------------
// Pass 3: out-projection + residual.  out = gamma * (wf @ AO + bf) + x
// ---------------------------------------------------------------------------
__global__ __launch_bounds__(256)
void out_proj_kernel(const float* __restrict__ AO,
                     const float* __restrict__ x,
                     const float* __restrict__ wf,
                     const float* __restrict__ bf_,
                     const float* __restrict__ gamma,
                     float* __restrict__ out)
{
  const int N = 4096;
  __shared__ float at[64][129];
  const int b  = blockIdx.x >> 5;
  const int n0 = (blockIdx.x & 31) * 128;
  const int t  = threadIdx.x;

  for (int i = t; i < 64 * 128; i += 256) {
    int c = i >> 7, nl = i & 127;
    at[c][nl] = AO[((size_t)(b * 64 + c)) * N + n0 + nl];
  }
  __syncthreads();

  const float g  = gamma[0];
  const int half = t >> 7;
  const int nl   = t & 127;
  const int n    = n0 + nl;
  const int o0   = half * 32;

  for (int o = o0; o < o0 + 32; ++o) {
    float acc = bf_[o];
    #pragma unroll 16
    for (int c = 0; c < 64; ++c)
      acc = fmaf(wf[o * 64 + c], at[c][nl], acc);
    size_t idx = ((size_t)(b * 64 + o)) * N + n;
    out[idx] = g * acc + x[idx];
  }
}

// ---------------------------------------------------------------------------
extern "C" void kernel_launch(void* const* d_in, const int* in_sizes, int n_in,
                              void* d_out, int out_size, void* d_ws, size_t ws_size,
                              hipStream_t stream) {
  (void)in_sizes; (void)n_in; (void)out_size; (void)ws_size;
  const float* x     = (const float*)d_in[0];
  const float* wq    = (const float*)d_in[1];
  const float* bq    = (const float*)d_in[2];
  const float* wk    = (const float*)d_in[3];
  const float* bk    = (const float*)d_in[4];
  const float* wv    = (const float*)d_in[5];
  const float* bv    = (const float*)d_in[6];
  const float* wf    = (const float*)d_in[7];
  const float* bf_   = (const float*)d_in[8];
  const float* gamma = (const float*)d_in[9];
  float* out = (float*)d_out;

  // workspace layout (16 MB total)
  char* ws = (char*)d_ws;
  unsigned short* Qb = (unsigned short*)(ws);                       // 2 MB
  unsigned short* Kb = (unsigned short*)(ws + (2u << 20));          // 2 MB
  unsigned short* Vt = (unsigned short*)(ws + (4u << 20));          // 4 MB
  float*          AO = (float*)(ws + (8u << 20));                   // 8 MB

  qkv_proj_kernel<<<256, 256, 0, stream>>>(x, wq, bq, wk, bk, wv, bv,
                                           Qb, Kb, Vt);
  flash_attn_kernel<<<512, 128, 0, stream>>>(Qb, Kb, Vt, AO);
  out_proj_kernel<<<256, 256, 0, stream>>>(AO, x, wf, bf_, gamma, out);
}
